// AR_Attention_22127671509571
// MI455X (gfx1250) — matrix operand order repaired and checked by reading, not compiled, not run on hardware
//
#include <hip/hip_runtime.h>
#include <hip/hip_bf16.h>

typedef __attribute__((ext_vector_type(16))) __bf16 v16bf;
typedef __attribute__((ext_vector_type(8)))  float  v8f;

__device__ __forceinline__ v8f wmma_bf16(v16bf a, v16bf b, v8f c) {
  return __builtin_amdgcn_wmma_f32_16x16x32_bf16(false, a, false, b, (short)0, c, false, false);
}

// ---------------------------------------------------------------------------
// K0: pack a row-major fp32 weight matrix (K=256 rows, N cols) into bf16 WMMA
// B-fragment order: frag = ntile*8 + ks, element layout [frag][lane][16] with
//   value = W[(ks*32 + (lane>>4)*16 + j) * N + ntile*16 + (lane&15)]
// so the GEMM hot loop loads each fragment as one contiguous 32B v16bf.
// ---------------------------------------------------------------------------
__global__ __launch_bounds__(32) void k_packw(const float* __restrict__ W,
                                              __bf16* __restrict__ P, int N) {
  const int frag  = blockIdx.x;          // ntile*8 + ks
  const int ntile = frag >> 3, ks = frag & 7;
  const int L = threadIdx.x;
  __bf16* dst = P + ((size_t)frag * 32 + L) * 16;
  const int kbase = ks * 32 + (L >> 4) * 16;
  const int n = ntile * 16 + (L & 15);
#pragma unroll
  for (int j = 0; j < 16; ++j) dst[j] = (__bf16)W[(size_t)(kbase + j) * N + n];
}

// ---------------------------------------------------------------------------
// K1: QKV projection GEMM.  M=12544 rows (b,p,pix windowed order), K=256, N=768.
// A rows gathered from image-layout x (window permute fused into addressing).
// B fragments come prepacked (bf16, fragment-contiguous).
// ---------------------------------------------------------------------------
__global__ __launch_bounds__(256) void k_qkv(const float* __restrict__ x,
                                             const __bf16* __restrict__ pw,
                                             const float* __restrict__ bqkv,
                                             __bf16* __restrict__ q,
                                             __bf16* __restrict__ k,
                                             __bf16* __restrict__ v,
                                             float* __restrict__ vimg) {
  const int wave = threadIdx.x >> 5, lane = threadIdx.x & 31;
  const int lh = lane >> 4, ln = lane & 15;
  const int mtile = blockIdx.x;              // 0..783
  const int ntile = blockIdx.y * 8 + wave;   // 0..47
  const int n = ntile * 16 + ln;

  const int m = mtile * 16 + ln;
  const int b_  = m / (49 * 64);
  const int rem = m % (49 * 64);
  const int p   = rem / 64;
  const int pix = rem % 64;
  const int Hg = (p / 7) * 8 + (pix >> 3);
  const int Wg = (p % 7) * 8 + (pix & 7);
  const float* arow = x + (size_t)((b_ * 56 + Hg) * 56 + Wg) * 256;

  const v16bf* pB = (const v16bf*)pw + (size_t)ntile * 8 * 32 + lane;

  v8f acc = {};
#pragma unroll
  for (int ks = 0; ks < 8; ++ks) {
    const int kb = ks * 32 + lh * 16;
    v16bf a;
#pragma unroll
    for (int j = 0; j < 8; ++j) {
      a[j]     = (__bf16)arow[kb + j];
      a[j + 8] = (__bf16)arow[kb + 8 + j];
    }
    acc = wmma_bf16(a, pB[ks * 32], acc);
  }
  const float bias = bqkv[n];
#pragma unroll
  for (int i = 0; i < 8; ++i) {
    const int mr = mtile * 16 + i + lh * 8;
    const float val = acc[i] + bias;
    if (n < 256) {
      q[(size_t)mr * 256 + n] = (__bf16)val;
    } else if (n < 512) {
      k[(size_t)mr * 256 + (n - 256)] = (__bf16)val;
    } else {
      v[(size_t)mr * 256 + (n - 512)] = (__bf16)val;
      const int bb = mr / (49 * 64);
      const int rr = mr % (49 * 64);
      const int pp = rr / 64, px = rr % 64;
      const int hg = (pp / 7) * 8 + (px >> 3);
      const int wg = (pp % 7) * 8 + (px & 7);
      vimg[(size_t)((bb * 56 + hg) * 56 + wg) * 256 + (n - 512)] = val;
    }
  }
}

// ---------------------------------------------------------------------------
// K2: per-window means of q and k  -> q_win, k_win (B*49, 256) fp32
// ---------------------------------------------------------------------------
__global__ __launch_bounds__(256) void k_means(const __bf16* __restrict__ q,
                                               const __bf16* __restrict__ k,
                                               float* __restrict__ qwin,
                                               float* __restrict__ kwin) {
  const int bp = blockIdx.x;       // 0..195
  const int c  = threadIdx.x;      // 0..255
  const __bf16* qb = q + (size_t)bp * 64 * 256 + c;
  const __bf16* kb = k + (size_t)bp * 64 * 256 + c;
  float sq = 0.f, sk = 0.f;
  for (int pix = 0; pix < 64; ++pix) {
    sq += (float)qb[(size_t)pix * 256];
    sk += (float)kb[(size_t)pix * 256];
  }
  qwin[(size_t)bp * 256 + c] = sq * (1.f / 64.f);
  kwin[(size_t)bp * 256 + c] = sk * (1.f / 64.f);
}

// ---------------------------------------------------------------------------
// K3: routing logits (49x49) + top-4 per (b,p).  Only the SET of indices
// matters (softmax+sum over concatenated keys is permutation invariant).
// ---------------------------------------------------------------------------
__global__ __launch_bounds__(64) void k_route(const float* __restrict__ qwin,
                                              const float* __restrict__ kwin,
                                              int* __restrict__ ridx) {
  const int b = blockIdx.x;  // 0..3
  const int p = blockIdx.y;  // 0..48
  __shared__ float sL[49];
  const int t = threadIdx.x;
  if (t < 49) {
    const float* qv = qwin + ((size_t)b * 49 + p) * 256;
    const float* kv = kwin + ((size_t)b * 49 + t) * 256;
    float d = 0.f;
    for (int c = 0; c < 256; ++c) d += qv[c] * kv[c];
    sL[t] = d;
  }
  __syncthreads();
  if (t == 0) {
    bool used[49];
    for (int i = 0; i < 49; ++i) used[i] = false;
    for (int s = 0; s < 4; ++s) {
      float best = -3.4e38f;
      int bi = 0;
      for (int i = 0; i < 49; ++i)
        if (!used[i] && sL[i] > best) { best = sL[i]; bi = i; }
      used[bi] = true;
      ridx[((size_t)b * 49 + p) * 4 + s] = bi;
    }
  }
}

// ---------------------------------------------------------------------------
// K4: LEPE depthwise 5x5 SAME conv on v_img + bias -> lepe (image layout)
// ---------------------------------------------------------------------------
__global__ __launch_bounds__(256) void k_lepe(const float* __restrict__ vimg,
                                              const float* __restrict__ w,
                                              const float* __restrict__ bias,
                                              float* __restrict__ lepe) {
  const int pixel = blockIdx.x;    // 0 .. 4*56*56-1
  const int c = threadIdx.x;
  const int b = pixel / (56 * 56);
  const int rem = pixel % (56 * 56);
  const int y = rem / 56, xx = rem % 56;
  float acc = bias[c];
#pragma unroll
  for (int dy = 0; dy < 5; ++dy) {
    const int yy = y + dy - 2;
    if (yy < 0 || yy >= 56) continue;
#pragma unroll
    for (int dx = 0; dx < 5; ++dx) {
      const int xz = xx + dx - 2;
      if (xz < 0 || xz >= 56) continue;
      acc += vimg[(size_t)((b * 56 + yy) * 56 + xz) * 256 + c] * w[(dy * 5 + dx) * 256 + c];
    }
  }
  lepe[(size_t)pixel * 256 + c] = acc;
}

// ---------------------------------------------------------------------------
// K5: routed attention.  One block per (b,p,head): 8 waves.
//   Stage:   V_sel transposed into LDS (sVT[d][key], coalesced copy)
//   Phase B: S = Q(64x32) @ K_sel^T(32x256) via WMMA into 64KB LDS (fp32)
//   Phase C: in-place row softmax (64 row threads), scale = 1/16
//   Phase D: O = P(64x256) @ V_sel(256x32) via WMMA, B frags from sVT (b128)
// ---------------------------------------------------------------------------
__global__ __launch_bounds__(256) void k_attn(const __bf16* __restrict__ q,
                                              const __bf16* __restrict__ k,
                                              const __bf16* __restrict__ v,
                                              const int* __restrict__ ridx,
                                              float* __restrict__ y) {
  __shared__ float  sS[64 * 256];    // 64KB: S, then P in place
  __shared__ __bf16 sVT[32 * 256];   // 16KB: V_sel transposed, d-major
  __shared__ int sWin[4];
  const int blk  = blockIdx.x;       // (b*49+p)*8 + head
  const int head = blk & 7;
  const int bp   = blk >> 3;
  const int b    = bp / 49, p = bp % 49;
  const int tid  = threadIdx.x, wave = tid >> 5, lane = tid & 31;
  const int lh = lane >> 4, ln = lane & 15;

  if (tid < 4) sWin[tid] = ridx[(size_t)bp * 4 + tid];
  __syncthreads();

  const __bf16* qbase = q + (size_t)bp * 64 * 256 + head * 32;
  const __bf16* kbase = k + (size_t)b * 49 * 64 * 256 + head * 32;
  const __bf16* vbase = v + (size_t)b * 49 * 64 * 256 + head * 32;

  // Stage V_sel^T: thread `tid` copies routed key row `tid` (64B coalesced read)
  {
    const __bf16* vrow = vbase + ((size_t)sWin[tid >> 6] * 64 + (tid & 63)) * 256;
#pragma unroll
    for (int d = 0; d < 32; ++d) sVT[d * 256 + tid] = vrow[d];
  }

  // Preload Q A-fragments for all 4 M-tiles (K=hd=32: one fragment each)
  v16bf aQ[4];
#pragma unroll
  for (int mt = 0; mt < 4; ++mt) {
    const __bf16* rp = qbase + (size_t)(mt * 16 + ln) * 256;
    const int kb = lh * 16;
    v16bf a;
#pragma unroll
    for (int j = 0; j < 8; ++j) { a[j] = rp[kb + j]; a[j + 8] = rp[kb + 8 + j]; }
    aQ[mt] = a;
  }

  // Phase B: each wave handles N-tiles {2*wave, 2*wave+1}, all 4 M-tiles
#pragma unroll
  for (int ni = 0; ni < 2; ++ni) {
    const int nt = wave * 2 + ni;
    const int r  = nt * 16 + ln;                       // routed key row 0..255
    const __bf16* krow = kbase + ((size_t)sWin[r >> 6] * 64 + (r & 63)) * 256;
    const v16bf bK = *(const v16bf*)(krow + lh * 16);  // 32B contiguous
#pragma unroll
    for (int mt = 0; mt < 4; ++mt) {
      v8f c = {};
      c = wmma_bf16(aQ[mt], bK, c);
#pragma unroll
      for (int i = 0; i < 8; ++i)
        sS[(mt * 16 + i + lh * 8) * 256 + nt * 16 + ln] = c[i];
    }
  }
  __syncthreads();

  // Phase C: softmax rows (scale applied to logits), in place fp32
  if (tid < 64) {
    float* rp = sS + (size_t)tid * 256;
    const float scale = 0.0625f;  // 256^-0.5
    float mx = -3.4e38f;
    for (int i = 0; i < 256; ++i) { const float vv = rp[i] * scale; mx = vv > mx ? vv : mx; }
    float sum = 0.f;
    for (int i = 0; i < 256; ++i) { const float e = __expf(rp[i] * scale - mx); rp[i] = e; sum += e; }
    const float inv = 1.f / sum;
    for (int i = 0; i < 256; ++i) rp[i] *= inv;
  }
  __syncthreads();

  // Phase D: O = P @ V.  wave -> (mt = wave>>1, nt = wave&1)
  const int mt = wave >> 1, nt = wave & 1;
  v8f acc = {};
#pragma unroll
  for (int ks = 0; ks < 8; ++ks) {
    const float* prow = sS + (size_t)(mt * 16 + ln) * 256 + ks * 32 + lh * 16;
    v16bf aP;
#pragma unroll
    for (int j = 0; j < 8; ++j) {
      aP[j]     = (__bf16)prow[j];
      aP[j + 8] = (__bf16)prow[8 + j];
    }
    // B fragment: lane n = d, K = key index -> contiguous in sVT (32B)
    const v16bf bV = *(const v16bf*)(sVT + (nt * 16 + ln) * 256 + ks * 32 + lh * 16);
    acc = wmma_bf16(aP, bV, acc);
  }
  const int wi = p / 7, wj = p % 7;
#pragma unroll
  for (int i = 0; i < 8; ++i) {
    const int pix = mt * 16 + i + lh * 8;
    const int hg = wi * 8 + (pix >> 3), wg = wj * 8 + (pix & 7);
    y[(size_t)((b * 56 + hg) * 56 + wg) * 256 + head * 32 + nt * 16 + ln] = acc[i];
  }
}

// ---------------------------------------------------------------------------
// K6: out = (y + lepe) @ Wo + bo.  M=12544 (image rows), K=256, N=256.
// ---------------------------------------------------------------------------
__global__ __launch_bounds__(256) void k_out(const float* __restrict__ y,
                                             const float* __restrict__ lepe,
                                             const __bf16* __restrict__ pwo,
                                             const float* __restrict__ bo,
                                             float* __restrict__ out) {
  const int wave = threadIdx.x >> 5, lane = threadIdx.x & 31;
  const int lh = lane >> 4, ln = lane & 15;
  const int mtile = blockIdx.x;              // 0..783
  const int ntile = blockIdx.y * 8 + wave;   // 0..15
  const int n = ntile * 16 + ln;
  const int m = mtile * 16 + ln;
  const float* yr = y + (size_t)m * 256;
  const float* lr = lepe + (size_t)m * 256;

  const v16bf* pB = (const v16bf*)pwo + (size_t)ntile * 8 * 32 + lane;

  v8f acc = {};
#pragma unroll
  for (int ks = 0; ks < 8; ++ks) {
    const int kb = ks * 32 + lh * 16;
    v16bf a;
#pragma unroll
    for (int j = 0; j < 8; ++j) {
      a[j]     = (__bf16)(yr[kb + j]      + lr[kb + j]);
      a[j + 8] = (__bf16)(yr[kb + 8 + j] + lr[kb + 8 + j]);
    }
    acc = wmma_bf16(a, pB[ks * 32], acc);
  }
  const float bias = bo[n];
#pragma unroll
  for (int i = 0; i < 8; ++i)
    out[(size_t)(mtile * 16 + i + lh * 8) * 256 + n] = acc[i] + bias;
}

// ---------------------------------------------------------------------------
extern "C" void kernel_launch(void* const* d_in, const int* in_sizes, int n_in,
                              void* d_out, int out_size, void* d_ws, size_t ws_size,
                              hipStream_t stream) {
  (void)in_sizes; (void)n_in; (void)out_size; (void)ws_size;
  const float* x      = (const float*)d_in[0];
  const float* Wqkv   = (const float*)d_in[1];
  const float* bqkv   = (const float*)d_in[2];
  const float* Wo     = (const float*)d_in[3];
  const float* bo     = (const float*)d_in[4];
  const float* lepe_w = (const float*)d_in[5];
  const float* lepe_b = (const float*)d_in[6];
  float* out = (float*)d_out;

  const size_t M = 12544;            // B*49*64 == B*56*56
  char* ws = (char*)d_ws;
  size_t off = 0;
  auto carve = [&](size_t bytes) { char* p = ws + off; off = (off + bytes + 255) & ~(size_t)255; return p; };

  __bf16* qb    = (__bf16*)carve(M * 256 * sizeof(__bf16));
  __bf16* kb    = (__bf16*)carve(M * 256 * sizeof(__bf16));
  __bf16* vb    = (__bf16*)carve(M * 256 * sizeof(__bf16));
  float*  vimg  = (float*)carve(M * 256 * sizeof(float));
  float*  lepe  = (float*)carve(M * 256 * sizeof(float));
  float*  ybuf  = (float*)carve(M * 256 * sizeof(float));
  float*  qwin  = (float*)carve(4 * 49 * 256 * sizeof(float));
  float*  kwin  = (float*)carve(4 * 49 * 256 * sizeof(float));
  int*    ridx  = (int*)carve(4 * 49 * 4 * sizeof(int));
  __bf16* pWqkv = (__bf16*)carve(256 * 768 * sizeof(__bf16));
  __bf16* pWo   = (__bf16*)carve(256 * 256 * sizeof(__bf16));

  k_packw<<<dim3(48 * 8),  32, 0, stream>>>(Wqkv, pWqkv, 768);
  k_packw<<<dim3(16 * 8),  32, 0, stream>>>(Wo,   pWo,   256);
  k_qkv  <<<dim3(784, 6), 256, 0, stream>>>(x, pWqkv, bqkv, qb, kb, vb, vimg);
  k_means<<<dim3(196),    256, 0, stream>>>(qb, kb, qwin, kwin);
  k_route<<<dim3(4, 49),   64, 0, stream>>>(qwin, kwin, ridx);
  k_lepe <<<dim3(12544),  256, 0, stream>>>(vimg, lepe_w, lepe_b, lepe);
  k_attn <<<dim3(1568),   256, 0, stream>>>(qb, kb, vb, ridx, ybuf);
  k_out  <<<dim3(784, 2), 256, 0, stream>>>(ybuf, lepe, pWo, bo, out);
}
